// CHMBlock_46737834115455
// MI455X (gfx1250) — compile-verified, matrix operand order checked by reading, not run on hardware
//
#include <hip/hip_runtime.h>
#include <hip/hip_bf16.h>

typedef _Float16 v16h __attribute__((ext_vector_type(16)));
typedef _Float16 v8h  __attribute__((ext_vector_type(8)));
typedef float    v8f  __attribute__((ext_vector_type(8)));

#define B_  2
#define D_  512
#define LC_ 2048
#define M_  4096
#define H_  8
#define HD_ 64
#define N_  16

// Load a 16x32 f16 WMMA A/B fragment. Row-major source, row stride ld (elems).
// Lane m<16 holds row m; halves 0..7 -> k = kbase+off..+7, halves 8..15 -> +16.
// off = 8*(lane/16). Two contiguous 16B chunks per lane.
static __device__ __forceinline__ v16h load_frag(const _Float16* base, int ld,
                                                 int row, int kbase, int lane) {
  const _Float16* p = base + (size_t)row * ld + kbase + ((lane >> 4) << 3);
  v8h lo = *(const v8h*)p;
  v8h hi = *(const v8h*)(p + 16);
  v16h r;
#pragma unroll
  for (int i = 0; i < 8; ++i) { r[i] = lo[i]; r[i + 8] = hi[i]; }
  return r;
}

static __device__ __forceinline__ float softplusf(float x) {
  return x > 20.f ? x : log1pf(__expf(x));
}

// ---------------- f32 -> f16 convert ----------------
__global__ void cvt16_k(const float* __restrict__ s, _Float16* __restrict__ d, int n) {
  int i = blockIdx.x * blockDim.x + threadIdx.x;
  if (i < n) d[i] = (_Float16)s[i];
}

// ------- LayerNorm over concat(feat_padded, text); wave per row; f16 out -------
__global__ void lnc_k(const float* __restrict__ feat, int Lfeat,
                      const float* __restrict__ text,
                      const float* __restrict__ g, const float* __restrict__ bt,
                      _Float16* __restrict__ out) {
  int lane = threadIdx.x & 31;
  int row  = __builtin_amdgcn_readfirstlane(
                 (int)((blockIdx.x * blockDim.x + threadIdx.x) >> 5));
  if (row >= B_ * LC_) return;
  int b = row >> 11, l = row & (LC_ - 1);
  const float* src = nullptr;
  if (l < 1024) { if (l < Lfeat) src = feat + ((size_t)b * Lfeat + l) * D_; }
  else           src = text + ((size_t)b * 1024 + (l - 1024)) * D_;
  float v[16]; float s = 0.f;
#pragma unroll
  for (int i = 0; i < 16; ++i) { float x = src ? src[lane + (i << 5)] : 0.f; v[i] = x; s += x; }
#pragma unroll
  for (int m = 16; m >= 1; m >>= 1) s += __shfl_xor(s, m, 32);
  float mean = s * (1.f / D_);
  float q = 0.f;
#pragma unroll
  for (int i = 0; i < 16; ++i) { float d0 = v[i] - mean; q += d0 * d0; }
#pragma unroll
  for (int m = 16; m >= 1; m >>= 1) q += __shfl_xor(q, m, 32);
  float r = rsqrtf(q * (1.f / D_) + 1e-5f);
#pragma unroll
  for (int i = 0; i < 16; ++i) {
    int c = lane + (i << 5);
    out[(size_t)row * D_ + c] = (_Float16)((v[i] - mean) * r * g[c] + bt[c]);
  }
}

// ------- y2 = y + LayerNorm(a); wave per row; f16 out -------
__global__ void addln_k(const float* __restrict__ y, const float* __restrict__ a,
                        const float* __restrict__ g, const float* __restrict__ bt,
                        _Float16* __restrict__ out) {
  int lane = threadIdx.x & 31;
  int row  = __builtin_amdgcn_readfirstlane(
                 (int)((blockIdx.x * blockDim.x + threadIdx.x) >> 5));
  if (row >= B_ * LC_) return;
  const float* src = a + (size_t)row * D_;
  float v[16]; float s = 0.f;
#pragma unroll
  for (int i = 0; i < 16; ++i) { float x = src[lane + (i << 5)]; v[i] = x; s += x; }
#pragma unroll
  for (int m = 16; m >= 1; m >>= 1) s += __shfl_xor(s, m, 32);
  float mean = s * (1.f / D_);
  float q = 0.f;
#pragma unroll
  for (int i = 0; i < 16; ++i) { float d0 = v[i] - mean; q += d0 * d0; }
#pragma unroll
  for (int m = 16; m >= 1; m >>= 1) q += __shfl_xor(q, m, 32);
  float r = rsqrtf(q * (1.f / D_) + 1e-5f);
#pragma unroll
  for (int i = 0; i < 16; ++i) {
    int c = lane + (i << 5);
    float ln = (v[i] - mean) * r * g[c] + bt[c];
    out[(size_t)row * D_ + c] = (_Float16)(y[(size_t)row * D_ + c] + ln);
  }
}

// ------- WMMA GEMM tile body: 16 x (NS*16) output per wave -------
template <int NS>
static __device__ __forceinline__ void gemm_tile(const _Float16* __restrict__ A, int lda,
                                                 const _Float16* __restrict__ W, int ldw,
                                                 const float* __restrict__ bias,
                                                 float* __restrict__ Cf,
                                                 _Float16* __restrict__ Ch,
                                                 int N, int K, int act,
                                                 int m0, int n0, int lane) {
  v8f acc[NS] = {};
  int arow = m0 + (lane & 15);
  const _Float16* ap = A + (size_t)arow * lda + ((lane >> 4) << 3);
  for (int kk = 0; kk < K; kk += 32) {
    __builtin_prefetch(ap + kk + 32, 0, 3);  // global_prefetch_b8, WGP scope
    v16h a = load_frag(A, lda, arow, kk, lane);
#pragma unroll
    for (int s = 0; s < NS; ++s) {
      v16h b = load_frag(W, ldw, n0 + (s << 4) + (lane & 15), kk, lane);
      acc[s] = __builtin_amdgcn_wmma_f32_16x16x32_f16(false, a, false, b,
                                                      (short)0, acc[s], false, false);
    }
  }
#pragma unroll
  for (int s = 0; s < NS; ++s) {
    int nc = n0 + (s << 4) + (lane & 15);
    float bv = bias ? bias[nc] : 0.f;
#pragma unroll
    for (int v = 0; v < 8; ++v) {
      int m = m0 + v + ((lane >> 4) << 3);
      float val = acc[s][v] + bv;
      if (act == 1) val = softplusf(val);
      size_t idx = (size_t)m * N + nc;
      if (Cf) Cf[idx] = val;
      if (Ch) Ch[idx] = (_Float16)val;
    }
  }
}

// C[M,N] = A[M,K] @ W[N,K]^T + bias ; wave computes a 16x64 (or 16x16) tile.
__global__ void gemm_k(const _Float16* __restrict__ A, int lda,
                       const _Float16* __restrict__ W, int ldw,
                       const float* __restrict__ bias,
                       float* __restrict__ Cf, _Float16* __restrict__ Ch,
                       int M, int N, int K, int act) {
  int lane = threadIdx.x & 31;
  int gw   = __builtin_amdgcn_readfirstlane(
                 (int)((blockIdx.x * blockDim.x + threadIdx.x) >> 5));
  int nt64 = (N + 63) >> 6;
  int mt = gw / nt64, nt = gw - mt * nt64;
  if (mt * 16 >= M) return;                 // scalar branch, EXEC untouched
  int m0 = mt << 4, n0 = nt << 6;
  int rem = (N - n0) >> 4;
  if (rem >= 4)
    gemm_tile<4>(A, lda, W, ldw, bias, Cf, Ch, N, K, act, m0, n0, lane);
  else
    gemm_tile<1>(A, lda, W, ldw, bias, Cf, Ch, N, K, act, m0, n0, lane);
}

// ------- selective-state scan: thread per (b,d,n); shuffle-reduce over n -------
__global__ void ssm_k(const float* __restrict__ x, const float* __restrict__ dt,
                      const float* __restrict__ Bm, const float* __restrict__ Cm,
                      const float* __restrict__ Alog, const float* __restrict__ Dsk,
                      float* __restrict__ y, _Float16* __restrict__ y16) {
  int gid = blockIdx.x * blockDim.x + threadIdx.x;
  if (gid >= B_ * D_ * N_) return;
  int n = gid & 15, d = (gid >> 4) & (D_ - 1), b = gid >> 13;
  float Av  = -__expf(Alog[d * N_ + n]);
  float dsk = Dsk[d];
  const float* xp = x  + (size_t)b * LC_ * D_ + d;
  const float* dp = dt + (size_t)b * LC_ * D_ + d;
  const float* Bp = Bm + (size_t)b * LC_ * N_ + n;
  const float* Cp = Cm + (size_t)b * LC_ * N_ + n;
  float h = 0.f;
  for (int t = 0; t < LC_; ++t) {
    float xv  = xp[(size_t)t * D_];
    float dtv = dp[(size_t)t * D_];
    float dA  = __expf(dtv * Av);
    h = dA * h + dtv * xv * Bp[(size_t)t * N_];
    float p = h * Cp[(size_t)t * N_];
    p += __shfl_xor(p, 8, 32); p += __shfl_xor(p, 4, 32);
    p += __shfl_xor(p, 2, 32); p += __shfl_xor(p, 1, 32);
    if (n == 0) {
      float yv = p + xv * dsk;
      size_t idx = ((size_t)b * LC_ + t) * D_ + d;
      y[idx] = yv; y16[idx] = (_Float16)yv;
    }
  }
}

// ------- transpose V into Vt[b][h][dim][l] (f16) -------
__global__ void vt_k(const _Float16* __restrict__ qkv, _Float16* __restrict__ Vt) {
  int gid = blockIdx.x * blockDim.x + threadIdx.x;
  if (gid >= B_ * H_ * HD_ * LC_) return;
  int l   = gid & (LC_ - 1);
  int dim = (gid >> 11) & 63;
  int h   = (gid >> 17) & 7;
  int b   = gid >> 20;
  Vt[gid] = qkv[((size_t)b * LC_ + l) * (3 * D_) + 2 * D_ + h * HD_ + dim];
}

// ------- flash attention: wave = (b,h,16-query tile) over all 2048 keys -------
__global__ void attn_k(const _Float16* __restrict__ qkv, const _Float16* __restrict__ Vt,
                       _Float16* __restrict__ o16) {
  __shared__ _Float16 lds[4][16][40];   // padded stride: conflict-free tr buffer
  int lane = threadIdx.x & 31;
  int wib  = __builtin_amdgcn_readfirstlane((int)(threadIdx.x >> 5));
  int gw   = __builtin_amdgcn_readfirstlane((int)(blockIdx.x * 4)) + wib;
  if (gw >= B_ * H_ * (LC_ / 16)) return;
  int qt = gw & 127, h = (gw >> 7) & 7, b = gw >> 10;
  const _Float16* qb = qkv + (size_t)b * LC_ * (3 * D_) + h * HD_;
  const _Float16* kb = qb + D_;
  const _Float16* vt = Vt + ((size_t)(b * H_ + h)) * HD_ * LC_;
  int q0 = qt << 4;
  int r15 = lane & 15;
  v16h qa = load_frag(qb, 3 * D_, q0 + r15, 0,  lane);
  v16h qc = load_frag(qb, 3 * D_, q0 + r15, 32, lane);
  v8f o[4] = {};
  float mrow[8], lrow[8];
#pragma unroll
  for (int v = 0; v < 8; ++v) { mrow[v] = -1e30f; lrow[v] = 0.f; }
  for (int kc = 0; kc < LC_; kc += 32) {
    v8f s[2];
#pragma unroll
    for (int jj = 0; jj < 2; ++jj) {
      int kr = kc + (jj << 4) + r15;
      v16h k0 = load_frag(kb, 3 * D_, kr, 0,  lane);
      v16h k1 = load_frag(kb, 3 * D_, kr, 32, lane);
      v8f c = {};
      c = __builtin_amdgcn_wmma_f32_16x16x32_f16(false, qa, false, k0, (short)0, c, false, false);
      c = __builtin_amdgcn_wmma_f32_16x16x32_f16(false, qc, false, k1, (short)0, c, false, false);
      s[jj] = c;
    }
    // online softmax; row stats live per-VGPR, broadcast within 16-lane groups
#pragma unroll
    for (int v = 0; v < 8; ++v) {
      float s0 = s[0][v] * 0.125f;   // 1/sqrt(64)
      float s1 = s[1][v] * 0.125f;
      float cm = fmaxf(s0, s1);
      cm = fmaxf(cm, __shfl_xor(cm, 8, 32));
      cm = fmaxf(cm, __shfl_xor(cm, 4, 32));
      cm = fmaxf(cm, __shfl_xor(cm, 2, 32));
      cm = fmaxf(cm, __shfl_xor(cm, 1, 32));
      float mn   = fmaxf(mrow[v], cm);
      float corr = __expf(mrow[v] - mn);
      mrow[v] = mn;
      float p0 = __expf(s0 - mn);
      float p1 = __expf(s1 - mn);
      int mr = v + ((lane >> 4) << 3);
      lds[wib][mr][r15]      = (_Float16)p0;   // C-layout -> plain [row][key]
      lds[wib][mr][r15 + 16] = (_Float16)p1;
      float rs = p0 + p1;
      rs += __shfl_xor(rs, 8, 32); rs += __shfl_xor(rs, 4, 32);
      rs += __shfl_xor(rs, 2, 32); rs += __shfl_xor(rs, 1, 32);
      lrow[v] = lrow[v] * corr + rs;
#pragma unroll
      for (int d = 0; d < 4; ++d) o[d][v] *= corr;
    }
    asm volatile("s_wait_dscnt 0" ::: "memory");
    v16h pf;  // reload P in A-fragment layout
    {
      int koff = (lane >> 4) << 3;
#pragma unroll
      for (int i = 0; i < 8; ++i) {
        pf[i]     = lds[wib][r15][koff + i];
        pf[i + 8] = lds[wib][r15][koff + 16 + i];
      }
    }
#pragma unroll
    for (int d = 0; d < 4; ++d) {
      v16h vf = load_frag(vt, LC_, (d << 4) + r15, kc, lane);
      o[d] = __builtin_amdgcn_wmma_f32_16x16x32_f16(false, pf, false, vf, (short)0, o[d], false, false);
    }
  }
#pragma unroll
  for (int d = 0; d < 4; ++d) {
#pragma unroll
    for (int v = 0; v < 8; ++v) {
      int m = v + ((lane >> 4) << 3);
      float val = o[d][v] / lrow[v];
      o16[((size_t)b * LC_ + q0 + m) * D_ + h * HD_ + (d << 4) + r15] = (_Float16)val;
    }
  }
}

// ------- mean over sequence -------
__global__ void mean_k(const float* __restrict__ o, float* __restrict__ mres) {
  int gid = blockIdx.x * blockDim.x + threadIdx.x;
  if (gid >= B_ * D_) return;
  int b = gid >> 9, d = gid & (D_ - 1);
  float s = 0.f;
  for (int t = 0; t < LC_; ++t) s += o[((size_t)b * LC_ + t) * D_ + d];
  mres[gid] = s * (1.f / LC_);
}

// ------- cls = LN((mean_a + mean_v) * 0.5) -------
__global__ void cls_k(const float* __restrict__ ma, const float* __restrict__ mv,
                      const float* __restrict__ g, const float* __restrict__ bt,
                      float* __restrict__ cls) {
  int b = blockIdx.x, lane = threadIdx.x;
  float v[16]; float s = 0.f;
#pragma unroll
  for (int i = 0; i < 16; ++i) {
    int c = lane + (i << 5);
    float x = (ma[b * D_ + c] + mv[b * D_ + c]) * 0.5f;
    v[i] = x; s += x;
  }
#pragma unroll
  for (int m = 16; m >= 1; m >>= 1) s += __shfl_xor(s, m, 32);
  float mean = s * (1.f / D_);
  float q = 0.f;
#pragma unroll
  for (int i = 0; i < 16; ++i) { float d0 = v[i] - mean; q += d0 * d0; }
#pragma unroll
  for (int m = 16; m >= 1; m >>= 1) q += __shfl_xor(q, m, 32);
  float r = rsqrtf(q * (1.f / D_) + 1e-5f);
#pragma unroll
  for (int i = 0; i < 16; ++i) {
    int c = lane + (i << 5);
    cls[b * D_ + c] = (v[i] - mean) * r * g[c] + bt[c];
  }
}

extern "C" void kernel_launch(void* const* d_in, const int* in_sizes, int n_in,
                              void* d_out, int out_size, void* d_ws, size_t ws_size,
                              hipStream_t stream) {
  (void)in_sizes; (void)n_in; (void)out_size; (void)ws_size;
  const float* text = (const float*)d_in[0];
  const float* audio = (const float*)d_in[1];
  const float* video = (const float*)d_in[2];
  const float* in_g = (const float*)d_in[3];
  const float* in_b = (const float*)d_in[4];
  const float* Wi   = (const float*)d_in[5];
  const float* bi   = (const float*)d_in[6];
  const float* Wdt  = (const float*)d_in[7];
  const float* bdt  = (const float*)d_in[8];
  const float* WB   = (const float*)d_in[9];
  const float* bB   = (const float*)d_in[10];
  const float* WC   = (const float*)d_in[11];
  const float* bC   = (const float*)d_in[12];
  const float* Alog = (const float*)d_in[13];
  const float* Dsk  = (const float*)d_in[14];
  const float* Wqkv = (const float*)d_in[15];
  const float* bqkv = (const float*)d_in[16];
  const float* Wo   = (const float*)d_in[17];
  const float* bo   = (const float*)d_in[18];
  const float* an_g = (const float*)d_in[19];
  const float* an_b = (const float*)d_in[20];
  const float* Wout = (const float*)d_in[21];
  const float* bout = (const float*)d_in[22];
  const float* on_g = (const float*)d_in[23];
  const float* on_b = (const float*)d_in[24];

  float* cls   = (float*)d_out;
  float* out_a = cls + B_ * D_;
  float* out_v = out_a + (size_t)B_ * LC_ * D_;

  char* ws = (char*)d_ws;
  size_t off = 0;
  auto alloc = [&](size_t bytes) -> char* {
    char* p = ws + off;
    off += (bytes + 255) & ~(size_t)255;
    return p;
  };
  const size_t MD = (size_t)M_ * D_;
  _Float16* wWi   = (_Float16*)alloc(262144 * 2);
  _Float16* wWdt  = (_Float16*)alloc(262144 * 2);
  _Float16* wWB   = (_Float16*)alloc(8192 * 2);
  _Float16* wWC   = (_Float16*)alloc(8192 * 2);
  _Float16* wWqkv = (_Float16*)alloc(786432 * 2);
  _Float16* wWo   = (_Float16*)alloc(262144 * 2);
  _Float16* wWout = (_Float16*)alloc(262144 * 2);
  _Float16* xln16 = (_Float16*)alloc(MD * 2);
  float*    xif   = (float*)   alloc(MD * 4);
  _Float16* xi16  = (_Float16*)alloc(MD * 2);
  float*    dtf   = (float*)   alloc(MD * 4);
  float*    Bmf   = (float*)   alloc((size_t)M_ * N_ * 4);
  float*    Cmf   = (float*)   alloc((size_t)M_ * N_ * 4);
  float*    yf    = (float*)   alloc(MD * 4);
  _Float16* y16   = (_Float16*)alloc(MD * 2);
  _Float16* qkv16 = (_Float16*)alloc((size_t)M_ * 3 * D_ * 2);
  _Float16* Vt16  = (_Float16*)alloc(MD * 2);
  _Float16* oo16  = (_Float16*)alloc(MD * 2);
  float*    af    = (float*)   alloc(MD * 4);
  _Float16* y2h   = (_Float16*)alloc(MD * 2);
  float*    ma    = (float*)   alloc(B_ * D_ * 4);
  float*    mv    = (float*)   alloc(B_ * D_ * 4);

  auto cvt = [&](const float* s, _Float16* d, int n) {
    cvt16_k<<<(n + 255) / 256, 256, 0, stream>>>(s, d, n);
  };
  cvt(Wi, wWi, 262144);
  cvt(Wdt, wWdt, 262144);
  cvt(WB, wWB, 8192);
  cvt(WC, wWC, 8192);
  cvt(Wqkv, wWqkv, 786432);
  cvt(Wo, wWo, 262144);
  cvt(Wout, wWout, 262144);

  auto gemm = [&](const _Float16* A, const _Float16* W, const float* bias,
                  float* Cf, _Float16* Ch, int Nn, int act) {
    int waves = (M_ / 16) * ((Nn + 63) / 64);
    gemm_k<<<(waves + 3) / 4, 128, 0, stream>>>(A, D_, W, D_, bias, Cf, Ch, M_, Nn, D_, act);
  };

  for (int br = 0; br < 2; ++br) {
    const float* feat = (br == 0) ? audio : video;
    int Lf            = (br == 0) ? 800 : 1024;
    float* outp       = (br == 0) ? out_a : out_v;
    float* mp         = (br == 0) ? ma : mv;

    lnc_k<<<512, 256, 0, stream>>>(feat, Lf, text, in_g, in_b, xln16);
    gemm(xln16, wWi,   bi,   xif, xi16, D_, 0);
    gemm(xi16,  wWdt,  bdt,  dtf, nullptr, D_, 1);     // softplus epilogue
    gemm(xi16,  wWB,   bB,   Bmf, nullptr, N_, 0);
    gemm(xi16,  wWC,   bC,   Cmf, nullptr, N_, 0);
    ssm_k<<<(B_ * D_ * N_) / 256, 256, 0, stream>>>(xif, dtf, Bmf, Cmf, Alog, Dsk, yf, y16);
    gemm(y16,   wWqkv, bqkv, nullptr, qkv16, 3 * D_, 0);
    vt_k<<<(B_ * H_ * HD_ * LC_) / 256, 256, 0, stream>>>(qkv16, Vt16);
    attn_k<<<(B_ * H_ * (LC_ / 16)) / 4, 128, 0, stream>>>(qkv16, Vt16, oo16);
    gemm(oo16,  wWo,   bo,   af,  nullptr, D_, 0);
    addln_k<<<512, 256, 0, stream>>>(yf, af, an_g, an_b, y2h);
    gemm(y2h,   wWout, bout, outp, nullptr, D_, 0);
    mean_k<<<(B_ * D_ + 255) / 256, 256, 0, stream>>>(outp, mp);
  }
  cls_k<<<2, 32, 0, stream>>>(ma, mv, on_g, on_b, cls);
}
